// RecurrentGNN_61984968016267
// MI455X (gfx1250) — matrix operand order repaired and checked by reading, not compiled, not run on hardware
//
#include <hip/hip_runtime.h>

typedef __bf16 bf16;
typedef __attribute__((ext_vector_type(16))) __bf16 v16bf;
typedef __attribute__((ext_vector_type(8)))  __bf16 v8bf;
typedef __attribute__((ext_vector_type(8)))  float  v8f;
typedef __attribute__((ext_vector_type(4)))  float  v4f;

#define DEVINL __device__ __forceinline__

constexpr int B_ = 4, N_ = 4096, E_ = 131072, I_ = 128, H_ = 128, F_ = 140;
constexpr int NT = 8; // 128 / 16 column tiles

// ---------------- WMMA + math helpers ----------------

DEVINL v8f wmma_bf16(v16bf a, v16bf b, v8f c) {
  return __builtin_amdgcn_wmma_f32_16x16x32_bf16(false, a, false, b, (short)0, c,
                                                 false, false);
}

DEVINL v8f splat8(float x) {
  v8f v;
#pragma unroll
  for (int i = 0; i < 8; ++i) v[i] = x;
  return v;
}

// branchless fast activations (no EXEC divergence, no libm calls)
DEVINL float fast_tanh(float x) {
  float t = __expf(2.0f * x);                    // inf for big x, 0 for very neg
  return 1.0f - 2.0f * __builtin_amdgcn_rcpf(t + 1.0f);
}
DEVINL float fast_sigm(float x) {
  return __builtin_amdgcn_rcpf(1.0f + __expf(-x));
}

// B-fragment: pre-swizzled in workspace; lane's 16 bf16 are contiguous (32B).
DEVINL v16bf wfrag(const bf16* __restrict__ W, int lane, int kt, int nt) {
  return *(const v16bf*)(W + (((size_t)(kt * NT + nt) * 32 + lane) * 16));
}

// A-fragment from a 16x128 bf16 row-major LDS tile.
DEVINL v16bf lds_a_frag(const bf16* buf, int lane, int kt) {
  const int m = lane & 15;
  const int base = kt * 32 + ((lane >> 4) & 1) * 8;
  v8bf lo = *(const v8bf*)(buf + m * 128 + base);
  v8bf hi = *(const v8bf*)(buf + m * 128 + base + 16);
  v16bf a;
#pragma unroll
  for (int j = 0; j < 8; ++j) { a[j] = lo[j]; a[j + 8] = hi[j]; }
  return a;
}

// A-fragment gathered directly from a global f32 row (with zero-pad past kmax).
DEVINL v16bf ga_frag(const float* __restrict__ row, int lane, int kt, int kmax) {
  const int base = kt * 32 + ((lane >> 4) & 1) * 8;
  v16bf a;
  if (base + 24 <= kmax) {
    v4f x0 = *(const v4f*)(row + base);
    v4f x1 = *(const v4f*)(row + base + 4);
    v4f y0 = *(const v4f*)(row + base + 16);
    v4f y1 = *(const v4f*)(row + base + 20);
#pragma unroll
    for (int j = 0; j < 4; ++j) {
      a[j] = (bf16)x0[j]; a[j + 4] = (bf16)x1[j];
      a[j + 8] = (bf16)y0[j]; a[j + 12] = (bf16)y1[j];
    }
  } else {
#pragma unroll
    for (int j = 0; j < 8; ++j) {
      int k0 = base + j, k1 = base + 16 + j;
      a[j]     = (bf16)((k0 < kmax) ? row[k0] : 0.0f);
      a[j + 8] = (bf16)((k1 < kmax) ? row[k1] : 0.0f);
    }
  }
  return a;
}

// Paired-nt matmul, A from registers. Two independent accumulator chains and
// B-fragment loads for kt+1 issued before the kt fragments are consumed, so
// the load latency overlaps the WMMAs instead of a load->wait0->wmma chain.
template <int KT>
DEVINL void mm_pair_rega(const v16bf* afr, const bf16* __restrict__ Wf, int lane,
                         int nt0, v8f& acc0, v8f& acc1) {
  v16bf b0 = wfrag(Wf, lane, 0, nt0);
  v16bf b1 = wfrag(Wf, lane, 0, nt0 + 1);
#pragma unroll
  for (int kt = 0; kt < KT; ++kt) {
    v16bf n0 = b0, n1 = b1;
    if (kt + 1 < KT) {
      n0 = wfrag(Wf, lane, kt + 1, nt0);
      n1 = wfrag(Wf, lane, kt + 1, nt0 + 1);
    }
    acc0 = wmma_bf16(afr[kt], b0, acc0);
    acc1 = wmma_bf16(afr[kt], b1, acc1);
    b0 = n0; b1 = n1;
  }
}

// Paired-nt matmul, A streamed from an LDS tile.
template <int KT>
DEVINL void mm_pair_lds(const bf16* abuf, const bf16* __restrict__ Wf, int lane,
                        int nt0, v8f& acc0, v8f& acc1) {
  v16bf b0 = wfrag(Wf, lane, 0, nt0);
  v16bf b1 = wfrag(Wf, lane, 0, nt0 + 1);
  v16bf a = lds_a_frag(abuf, lane, 0);
#pragma unroll
  for (int kt = 0; kt < KT; ++kt) {
    v16bf n0 = b0, n1 = b1, an = a;
    if (kt + 1 < KT) {
      n0 = wfrag(Wf, lane, kt + 1, nt0);
      n1 = wfrag(Wf, lane, kt + 1, nt0 + 1);
      an = lds_a_frag(abuf, lane, kt + 1);
    }
    acc0 = wmma_bf16(a, b0, acc0);
    acc1 = wmma_bf16(a, b1, acc1);
    b0 = n0; b1 = n1; a = an;
  }
}

// ---------------- weight prep: f32 [K][128] -> bf16 B-fragment layout ----------------

__global__ void prep_w_kernel(const float* __restrict__ src, bf16* __restrict__ dst,
                              int Ksrc, int Ktiles) {
  int total = Ktiles * NT * 512;
  for (int i = blockIdx.x * blockDim.x + threadIdx.x; i < total;
       i += gridDim.x * blockDim.x) {
    int j = i & 15;
    int lane = (i >> 4) & 31;
    int fr = i >> 9;
    int nt = fr % NT, kt = fr / NT;
    int n = (lane & 15) + nt * 16;
    int k = kt * 32 + (j >> 3) * 16 + ((lane >> 4) & 1) * 8 + (j & 7);
    dst[i] = (bf16)((k < Ksrc) ? src[k * 128 + n] : 0.0f);
  }
}

// ---------------- degree counts ----------------

__global__ void count_kernel(const int* __restrict__ recv, float* __restrict__ cnt) {
  int e = blockIdx.x * blockDim.x + threadIdx.x;
  if (e < E_) atomicAdd(&cnt[recv[e]], 1.0f);
}

// ---------------- edge message MLP ----------------

__global__ void __launch_bounds__(256) msg_edge_kernel(
    const float* __restrict__ hidden, const int* __restrict__ recv,
    const int* __restrict__ send, const bf16* __restrict__ w1f,
    const float* __restrict__ b1, const bf16* __restrict__ w2f,
    const float* __restrict__ b2, float* __restrict__ hid_sum) {
  __shared__ bf16 a2s[8][16 * 128];
  const int wave = threadIdx.x >> 5, lane = threadIdx.x & 31;
  const int tile = blockIdx.x * 8 + wave;
  const int b = blockIdx.y;
  const int e0 = tile * 16;
  const int mrow = lane & 15;
  const int half = (lane >> 4) & 1;
  bf16* a2 = &a2s[wave][0];

  const int er = e0 + mrow;
  const float* rowR = hidden + ((size_t)b * N_ + recv[er]) * H_;
  const float* rowS = hidden + ((size_t)b * N_ + send[er]) * H_;

  v16bf afr[8];
#pragma unroll
  for (int kt = 0; kt < 4; ++kt) afr[kt] = ga_frag(rowR, lane, kt, H_);
#pragma unroll
  for (int kt = 0; kt < 4; ++kt) afr[4 + kt] = ga_frag(rowS, lane, kt, H_);

  for (int nt = 0; nt < NT; nt += 2) {
    v8f acc0 = splat8(b1[nt * 16 + mrow]);
    v8f acc1 = splat8(b1[(nt + 1) * 16 + mrow]);
    mm_pair_rega<8>(afr, w1f, lane, nt, acc0, acc1);
    const int n0 = nt * 16 + mrow, n1 = n0 + 16;
#pragma unroll
    for (int r = 0; r < 8; ++r) {
      a2[(half * 8 + r) * 128 + n0] = (bf16)fast_tanh(acc0[r]);
      a2[(half * 8 + r) * 128 + n1] = (bf16)fast_tanh(acc1[r]);
    }
  }

  int scat[8];
#pragma unroll
  for (int r = 0; r < 8; ++r) scat[r] = recv[e0 + half * 8 + r];

  for (int nt = 0; nt < NT; nt += 2) {
    v8f acc0 = splat8(b2[nt * 16 + mrow]);
    v8f acc1 = splat8(b2[(nt + 1) * 16 + mrow]);
    mm_pair_lds<4>(a2, w2f, lane, nt, acc0, acc1);
    const int n0 = nt * 16 + mrow, n1 = n0 + 16;
#pragma unroll
    for (int r = 0; r < 8; ++r) {
      size_t base = ((size_t)b * N_ + scat[r]) * H_;
      atomicAdd(&hid_sum[base + n0], fast_tanh(acc0[r]));
      atomicAdd(&hid_sum[base + n1], fast_tanh(acc1[r]));
    }
  }
}

// ---------------- edge presence MLP ----------------

__global__ void __launch_bounds__(256) pm_edge_kernel(
    const float* __restrict__ edge_attr, const int* __restrict__ recv,
    const bf16* __restrict__ w1f, const float* __restrict__ b1,
    const bf16* __restrict__ w2f, const float* __restrict__ b2,
    float* __restrict__ pre_sum) {
  __shared__ bf16 a2s[8][16 * 128];
  const int wave = threadIdx.x >> 5, lane = threadIdx.x & 31;
  const int tile = blockIdx.x * 8 + wave;
  const int b = blockIdx.y;
  const int e0 = tile * 16;
  const int mrow = lane & 15;
  const int half = (lane >> 4) & 1;
  bf16* a2 = &a2s[wave][0];

  const float* rowE = edge_attr + ((size_t)b * E_ + e0 + mrow) * F_;
  __builtin_prefetch(rowE + (size_t)128 * F_, 0, 1);  // stream-ahead for HBM data

  v16bf afr[5];
#pragma unroll
  for (int kt = 0; kt < 5; ++kt) afr[kt] = ga_frag(rowE, lane, kt, F_);

  for (int nt = 0; nt < NT; nt += 2) {
    v8f acc0 = splat8(b1[nt * 16 + mrow]);
    v8f acc1 = splat8(b1[(nt + 1) * 16 + mrow]);
    mm_pair_rega<5>(afr, w1f, lane, nt, acc0, acc1);
    const int n0 = nt * 16 + mrow, n1 = n0 + 16;
#pragma unroll
    for (int r = 0; r < 8; ++r) {
      a2[(half * 8 + r) * 128 + n0] = (bf16)fmaxf(acc0[r], 0.0f);
      a2[(half * 8 + r) * 128 + n1] = (bf16)fmaxf(acc1[r], 0.0f);
    }
  }

  int scat[8];
#pragma unroll
  for (int r = 0; r < 8; ++r) scat[r] = recv[e0 + half * 8 + r];

  for (int nt = 0; nt < NT; nt += 2) {
    v8f acc0 = splat8(b2[nt * 16 + mrow]);
    v8f acc1 = splat8(b2[(nt + 1) * 16 + mrow]);
    mm_pair_lds<4>(a2, w2f, lane, nt, acc0, acc1);
    const int n0 = nt * 16 + mrow, n1 = n0 + 16;
#pragma unroll
    for (int r = 0; r < 8; ++r) {
      size_t base = ((size_t)b * N_ + scat[r]) * H_;
      atomicAdd(&pre_sum[base + n0], fmaxf(acc0[r], 0.0f));
      atomicAdd(&pre_sum[base + n1], fmaxf(acc1[r], 0.0f));
    }
  }
}

// ---------------- fused node kernel ----------------

__global__ void __launch_bounds__(128) node_kernel(
    const float* __restrict__ inputs, const float* __restrict__ hidden,
    const float* __restrict__ hid_sum, const float* __restrict__ pre_sum,
    const float* __restrict__ cnt,
    const bf16* __restrict__ res_w1f, const float* __restrict__ res_b1,
    const bf16* __restrict__ res_w2f, const float* __restrict__ res_b2,
    const bf16* __restrict__ ir_wf, const float* __restrict__ ir_b,
    const bf16* __restrict__ ii_wf, const float* __restrict__ ii_b,
    const bf16* __restrict__ in_wf, const float* __restrict__ in_b,
    const bf16* __restrict__ hr_wf, const bf16* __restrict__ hi_wf,
    const bf16* __restrict__ hh_wf,
    const bf16* __restrict__ out_w1f, const float* __restrict__ out_b1,
    const bf16* __restrict__ out_w2f, const float* __restrict__ out_b2,
    const bf16* __restrict__ out_w3f, const float* __restrict__ out_b3,
    float* __restrict__ out_pred, float* __restrict__ out_hidden) {
  __shared__ bf16 smem[4][3][16 * 128];
  const int wave = threadIdx.x >> 5, lane = threadIdx.x & 31;
  const int mrow = lane & 15, half = (lane >> 4) & 1;
  const int row0 = (blockIdx.x * 4 + wave) * 16;  // row index into flattened B*N
  bf16* bufA = &smem[wave][0][0];
  bf16* bufB = &smem[wave][1][0];
  bf16* bufC = &smem[wave][2][0];

  // stage input rows -> bufA (bf16)
  for (int m = 0; m < 16; ++m) {
    const float* row = inputs + (size_t)(row0 + m) * I_;
    v4f x = *(const v4f*)(row + lane * 4);
#pragma unroll
    for (int t = 0; t < 4; ++t) bufA[m * 128 + lane * 4 + t] = (bf16)x[t];
  }

  // res1 = relu(inputs @ res_w1 + b1) -> bufB
  for (int nt = 0; nt < NT; nt += 2) {
    v8f a0 = splat8(res_b1[nt * 16 + mrow]);
    v8f a1 = splat8(res_b1[(nt + 1) * 16 + mrow]);
    mm_pair_lds<4>(bufA, res_w1f, lane, nt, a0, a1);
    const int n0 = nt * 16 + mrow, n1 = n0 + 16;
#pragma unroll
    for (int r = 0; r < 8; ++r) {
      bufB[(half * 8 + r) * 128 + n0] = (bf16)fmaxf(a0[r], 0.0f);
      bufB[(half * 8 + r) * 128 + n1] = (bf16)fmaxf(a1[r], 0.0f);
    }
  }

  // present = relu(res1 @ res_w2 + b2) + pre_sum/max(cnt,1) -> bufA
  for (int nt = 0; nt < NT; nt += 2) {
    v8f a0 = splat8(res_b2[nt * 16 + mrow]);
    v8f a1 = splat8(res_b2[(nt + 1) * 16 + mrow]);
    mm_pair_lds<4>(bufB, res_w2f, lane, nt, a0, a1);
    const int n0 = nt * 16 + mrow, n1 = n0 + 16;
#pragma unroll
    for (int r = 0; r < 8; ++r) {
      int g = row0 + half * 8 + r;
      float ic = __builtin_amdgcn_rcpf(fmaxf(cnt[g & (N_ - 1)], 1.0f));
      float v0 = fmaxf(a0[r], 0.0f) + pre_sum[(size_t)g * H_ + n0] * ic;
      float v1 = fmaxf(a1[r], 0.0f) + pre_sum[(size_t)g * H_ + n1] * ic;
      bufA[(half * 8 + r) * 128 + n0] = (bf16)v0;
      bufA[(half * 8 + r) * 128 + n1] = (bf16)v1;
    }
  }

  // stage hidden_node_emb = hid_sum/max(cnt,1) -> bufB
  for (int m = 0; m < 16; ++m) {
    int g = row0 + m;
    float ic = __builtin_amdgcn_rcpf(fmaxf(cnt[g & (N_ - 1)], 1.0f));
    const float* row = hid_sum + (size_t)g * H_;
    v4f x = *(const v4f*)(row + lane * 4);
#pragma unroll
    for (int t = 0; t < 4; ++t) bufB[m * 128 + lane * 4 + t] = (bf16)(x[t] * ic);
  }

  // GRU: four WMMA accumulator chains (six WMMAs per k-step) per n-tile
  for (int nt = 0; nt < NT; ++nt) {
    const int n = nt * 16 + mrow;
    v8f rg = splat8(ir_b[n]);
    v8f ig = splat8(ii_b[n]);
    v8f ng = splat8(in_b[n]);
    v8f hg = splat8(0.0f);
#pragma unroll
    for (int kt = 0; kt < 4; ++kt) {
      v16bf ap = lds_a_frag(bufA, lane, kt);  // present
      v16bf ah = lds_a_frag(bufB, lane, kt);  // hidden emb
      rg = wmma_bf16(ap, wfrag(ir_wf, lane, kt, nt), rg);
      rg = wmma_bf16(ah, wfrag(hr_wf, lane, kt, nt), rg);
      ig = wmma_bf16(ap, wfrag(ii_wf, lane, kt, nt), ig);
      ig = wmma_bf16(ah, wfrag(hi_wf, lane, kt, nt), ig);
      ng = wmma_bf16(ap, wfrag(in_wf, lane, kt, nt), ng);
      hg = wmma_bf16(ah, wfrag(hh_wf, lane, kt, nt), hg);
    }
#pragma unroll
    for (int r = 0; r < 8; ++r) {
      int g = row0 + half * 8 + r;
      float rv = fast_sigm(rg[r]);
      float iv = fast_sigm(ig[r]);
      float nv = fast_tanh(ng[r] + rv * hg[r]);
      float hold = hidden[(size_t)g * H_ + n];
      float nh = (1.0f - iv) * nv + iv * hold;
      out_hidden[(size_t)g * H_ + n] = nh;
      bufC[(half * 8 + r) * 128 + n] = (bf16)nh;
    }
  }

  // o1 = relu(newh @ out_w1 + b1) -> bufA
  for (int nt = 0; nt < NT; nt += 2) {
    v8f a0 = splat8(out_b1[nt * 16 + mrow]);
    v8f a1 = splat8(out_b1[(nt + 1) * 16 + mrow]);
    mm_pair_lds<4>(bufC, out_w1f, lane, nt, a0, a1);
    const int n0 = nt * 16 + mrow, n1 = n0 + 16;
#pragma unroll
    for (int r = 0; r < 8; ++r) {
      bufA[(half * 8 + r) * 128 + n0] = (bf16)fmaxf(a0[r], 0.0f);
      bufA[(half * 8 + r) * 128 + n1] = (bf16)fmaxf(a1[r], 0.0f);
    }
  }
  // o2 = relu(o1 @ out_w2 + b2) -> bufB
  for (int nt = 0; nt < NT; nt += 2) {
    v8f a0 = splat8(out_b2[nt * 16 + mrow]);
    v8f a1 = splat8(out_b2[(nt + 1) * 16 + mrow]);
    mm_pair_lds<4>(bufA, out_w2f, lane, nt, a0, a1);
    const int n0 = nt * 16 + mrow, n1 = n0 + 16;
#pragma unroll
    for (int r = 0; r < 8; ++r) {
      bufB[(half * 8 + r) * 128 + n0] = (bf16)fmaxf(a0[r], 0.0f);
      bufB[(half * 8 + r) * 128 + n1] = (bf16)fmaxf(a1[r], 0.0f);
    }
  }
  // pred = o2 @ out_w3 + b3 -> out_pred
  for (int nt = 0; nt < NT; nt += 2) {
    v8f a0 = splat8(out_b3[nt * 16 + mrow]);
    v8f a1 = splat8(out_b3[(nt + 1) * 16 + mrow]);
    mm_pair_lds<4>(bufB, out_w3f, lane, nt, a0, a1);
    const int n0 = nt * 16 + mrow, n1 = n0 + 16;
#pragma unroll
    for (int r = 0; r < 8; ++r) {
      int g = row0 + half * 8 + r;
      out_pred[(size_t)g * I_ + n0] = a0[r];
      out_pred[(size_t)g * I_ + n1] = a1[r];
    }
  }
}

// ---------------- host launch ----------------

extern "C" void kernel_launch(void* const* d_in, const int* in_sizes, int n_in,
                              void* d_out, int out_size, void* d_ws, size_t ws_size,
                              hipStream_t stream) {
  (void)in_sizes; (void)n_in; (void)out_size; (void)ws_size;

  const float* inputs    = (const float*)d_in[0];
  const float* edge_attr = (const float*)d_in[1];
  const int*   send_e    = (const int*)d_in[2];
  const int*   recv_e    = (const int*)d_in[3];
  const float* hidden    = (const float*)d_in[4];
  const float* msg_w1 = (const float*)d_in[5];  const float* msg_b1 = (const float*)d_in[6];
  const float* msg_w2 = (const float*)d_in[7];  const float* msg_b2 = (const float*)d_in[8];
  const float* pm_w1  = (const float*)d_in[9];  const float* pm_b1  = (const float*)d_in[10];
  const float* pm_w2  = (const float*)d_in[11]; const float* pm_b2  = (const float*)d_in[12];
  const float* res_w1 = (const float*)d_in[13]; const float* res_b1 = (const float*)d_in[14];
  const float* res_w2 = (const float*)d_in[15]; const float* res_b2 = (const float*)d_in[16];
  const float* ir_w = (const float*)d_in[17];   const float* ir_b = (const float*)d_in[18];
  const float* ii_w = (const float*)d_in[19];   const float* ii_b = (const float*)d_in[20];
  const float* in_w = (const float*)d_in[21];   const float* in_b = (const float*)d_in[22];
  const float* hr_w = (const float*)d_in[23];
  const float* hi_w = (const float*)d_in[24];
  const float* hh_w = (const float*)d_in[25];
  const float* out_w1 = (const float*)d_in[26]; const float* out_b1 = (const float*)d_in[27];
  const float* out_w2 = (const float*)d_in[28]; const float* out_b2 = (const float*)d_in[29];
  const float* out_w3 = (const float*)d_in[30]; const float* out_b3 = (const float*)d_in[31];

  float* out_pred   = (float*)d_out;
  float* out_hidden = out_pred + (size_t)B_ * N_ * I_;

  char* p = (char*)d_ws;
  auto take = [&](size_t bytes) -> char* {
    char* r = p;
    p += (bytes + 255) & ~(size_t)255;
    return r;
  };
  float* hid_sum = (float*)take((size_t)B_ * N_ * H_ * sizeof(float));
  float* pre_sum = (float*)take((size_t)B_ * N_ * H_ * sizeof(float));
  float* cnt     = (float*)take((size_t)N_ * sizeof(float));
  auto takeW = [&](int ktiles) -> bf16* {
    return (bf16*)take((size_t)ktiles * NT * 512 * sizeof(bf16));
  };
  bf16* msg_w1f = takeW(8); bf16* msg_w2f = takeW(4);
  bf16* pm_w1f  = takeW(5); bf16* pm_w2f  = takeW(4);
  bf16* res_w1f = takeW(4); bf16* res_w2f = takeW(4);
  bf16* ir_wf = takeW(4);   bf16* ii_wf = takeW(4);   bf16* in_wf = takeW(4);
  bf16* hr_wf = takeW(4);   bf16* hi_wf = takeW(4);   bf16* hh_wf = takeW(4);
  bf16* out_w1f = takeW(4); bf16* out_w2f = takeW(4); bf16* out_w3f = takeW(4);

  hipMemsetAsync(hid_sum, 0, (size_t)B_ * N_ * H_ * sizeof(float), stream);
  hipMemsetAsync(pre_sum, 0, (size_t)B_ * N_ * H_ * sizeof(float), stream);
  hipMemsetAsync(cnt, 0, (size_t)N_ * sizeof(float), stream);

  auto prep = [&](const float* s, bf16* d, int Ksrc, int Kt) {
    prep_w_kernel<<<32, 256, 0, stream>>>(s, d, Ksrc, Kt);
  };
  prep(msg_w1, msg_w1f, 256, 8); prep(msg_w2, msg_w2f, 128, 4);
  prep(pm_w1, pm_w1f, 140, 5);   prep(pm_w2, pm_w2f, 128, 4);
  prep(res_w1, res_w1f, 128, 4); prep(res_w2, res_w2f, 128, 4);
  prep(ir_w, ir_wf, 128, 4); prep(ii_w, ii_wf, 128, 4); prep(in_w, in_wf, 128, 4);
  prep(hr_w, hr_wf, 128, 4); prep(hi_w, hi_wf, 128, 4); prep(hh_w, hh_wf, 128, 4);
  prep(out_w1, out_w1f, 128, 4); prep(out_w2, out_w2f, 128, 4);
  prep(out_w3, out_w3f, 128, 4);

  count_kernel<<<E_ / 256, 256, 0, stream>>>(recv_e, cnt);

  dim3 eg(E_ / (16 * 8), B_);
  msg_edge_kernel<<<eg, 256, 0, stream>>>(hidden, recv_e, send_e, msg_w1f, msg_b1,
                                          msg_w2f, msg_b2, hid_sum);
  pm_edge_kernel<<<eg, 256, 0, stream>>>(edge_attr, recv_e, pm_w1f, pm_b1, pm_w2f,
                                         pm_b2, pre_sum);

  node_kernel<<<(B_ * N_) / (16 * 4), 128, 0, stream>>>(
      inputs, hidden, hid_sum, pre_sum, cnt,
      res_w1f, res_b1, res_w2f, res_b2,
      ir_wf, ir_b, ii_wf, ii_b, in_wf, in_b,
      hr_wf, hi_wf, hh_wf,
      out_w1f, out_b1, out_w2f, out_b2, out_w3f, out_b3,
      out_pred, out_hidden);
}
